// MCQuantiles_16423954939968
// MI455X (gfx1250) — matrix operand order repaired and checked by reading, not compiled, not run on hardware
//
#include <hip/hip_runtime.h>

// ---------------------------------------------------------------------------
// CDNA5 (gfx1250) implementation of the 3-compartment SNN reference.
//   Phase A: time-batched projections  apical=te@Wa^T (M=16384), basal=se@Wb^T
//            (M=512), K=3136, N=512  -> bf16 WMMA GEMM, double-buffered LDS.
//   Phase B: ma/mb are linear IIR filters (no reset) -> precompute
//            drive[t] = ma_t + mb_t elementwise, in place.
//   Phase C: persistent sequential kernel over T=8; ms in LDS, ml + output
//            accumulator live in the WMMA C-fragment registers; spikes go
//            through LDS (bf16) into v_wmma for the W1 / W2 GEMMs; cross-wave
//            reduction with ds_add_f32 atomics.
// ---------------------------------------------------------------------------

typedef __attribute__((ext_vector_type(16))) __bf16 v16bf;
typedef __attribute__((ext_vector_type(8)))  float  v8f;

union Frag {
  v16bf        v;
  unsigned int u[8];
};

#define TAU_INV 0.5f
#define TT   8
#define SS   32
#define FF   512
#define LL   18
#define LPAD 32
#define NROW 2048        // B*S

__device__ __forceinline__ unsigned short f2bf(float f) {
  unsigned int u = __float_as_uint(f);          // round-to-nearest-even bf16
  u += 0x7FFFu + ((u >> 16) & 1u);
  return (unsigned short)(u >> 16);
}
__device__ __forceinline__ unsigned int pk2(float lo, float hi) {
  return (unsigned int)f2bf(lo) | ((unsigned int)f2bf(hi) << 16);
}

// ---------------------------------------------------------------------------
// Phase A: C[M,N] = A[M,K] @ W[N,K]^T, fp32 in, bf16 WMMA, fp32 out.
// Block tile 128x128, 8 waves, each wave 64x32 (4x2 WMMA 16x16 tiles).
// Double-buffered LDS: global loads for tile k+1 issue before the WMMAs of
// tile k; one barrier per K-step.
// ---------------------------------------------------------------------------
#define BM 128
#define BN 128
#define BK 32
#define LDA 34    // ldsA row stride (bf16 elems), padded vs 32
#define LDB 132   // ldsB row stride (bf16 elems), padded vs 128

__global__ __launch_bounds__(256) void gemm_f32_bf16(
    const float* __restrict__ A, const float* __restrict__ W,
    float* __restrict__ C, int M, int N, int K)
{
  __shared__ unsigned short ldsA[2][BM * LDA];  // [buf][m][k]
  __shared__ unsigned short ldsB[2][BK * LDB];  // [buf][k][n] (transposed stage)

  const int tid  = threadIdx.x;
  const int wave = tid >> 5;
  const int lane = tid & 31;
  const int lm   = lane & 15;
  const int lh   = lane >> 4;
  const int blockM = blockIdx.y * BM;
  const int blockN = blockIdx.x * BN;
  const int waveM  = (wave >> 2) * 64;
  const int waveN  = (wave & 3) * 32;

  const int arow  = tid >> 1;          // 0..127
  const int akoff = (tid & 1) * 16;    // 0 or 16

  const v8f vzero = {0.f,0.f,0.f,0.f,0.f,0.f,0.f,0.f};
  v8f acc[4][2];
#pragma unroll
  for (int i = 0; i < 4; ++i)
#pragma unroll
    for (int j = 0; j < 2; ++j) acc[i][j] = vzero;

  const float* aptr = A + (size_t)(blockM + arow) * K + akoff;
  const float* wptr = W + (size_t)(blockN + arow) * K + akoff;

  float4 a0, a1, a2, a3, w0, w1, w2, w3;
  // ---- preload tile 0 and stage into buffer 0
  {
    const float4* pa = (const float4*)(aptr);
    a0 = pa[0]; a1 = pa[1]; a2 = pa[2]; a3 = pa[3];
    const float4* pw = (const float4*)(wptr);
    w0 = pw[0]; w1 = pw[1]; w2 = pw[2]; w3 = pw[3];
    unsigned int* d = (unsigned int*)&ldsA[0][arow * LDA + akoff];
    d[0] = pk2(a0.x, a0.y); d[1] = pk2(a0.z, a0.w);
    d[2] = pk2(a1.x, a1.y); d[3] = pk2(a1.z, a1.w);
    d[4] = pk2(a2.x, a2.y); d[5] = pk2(a2.z, a2.w);
    d[6] = pk2(a3.x, a3.y); d[7] = pk2(a3.z, a3.w);
    unsigned short* db = &ldsB[0][akoff * LDB + arow];
    db[ 0*LDB] = f2bf(w0.x); db[ 1*LDB] = f2bf(w0.y);
    db[ 2*LDB] = f2bf(w0.z); db[ 3*LDB] = f2bf(w0.w);
    db[ 4*LDB] = f2bf(w1.x); db[ 5*LDB] = f2bf(w1.y);
    db[ 6*LDB] = f2bf(w1.z); db[ 7*LDB] = f2bf(w1.w);
    db[ 8*LDB] = f2bf(w2.x); db[ 9*LDB] = f2bf(w2.y);
    db[10*LDB] = f2bf(w2.z); db[11*LDB] = f2bf(w2.w);
    db[12*LDB] = f2bf(w3.x); db[13*LDB] = f2bf(w3.y);
    db[14*LDB] = f2bf(w3.z); db[15*LDB] = f2bf(w3.w);
  }
  __syncthreads();

  for (int kt = 0; kt < K; kt += BK) {
    const int  cur  = (kt >> 5) & 1;
    const bool more = (kt + BK) < K;

    // ---- issue global loads for tile k+1 (latency hidden behind WMMAs)
    if (more) {
      const float4* pa = (const float4*)(aptr + kt + BK);
      a0 = pa[0]; a1 = pa[1]; a2 = pa[2]; a3 = pa[3];
      const float4* pw = (const float4*)(wptr + kt + BK);
      w0 = pw[0]; w1 = pw[1]; w2 = pw[2]; w3 = pw[3];
      if (kt + 2 * BK < K) {           // gfx1250 global_prefetch_b8, near-cache
        __builtin_prefetch(aptr + kt + 2 * BK, 0, 3);
        __builtin_prefetch(wptr + kt + 2 * BK, 0, 3);
      }
    }

    // ---- fragments + WMMA from buffer `cur`
    Frag af[4], bfr[2];
#pragma unroll
    for (int ms = 0; ms < 4; ++ms) {
      const unsigned short* src = &ldsA[cur][(waveM + ms*16 + lm) * LDA];
#pragma unroll
      for (int r = 0; r < 8; ++r) {
        const int kk = ((r & 4) << 2) + lh * 8 + (r & 3) * 2; // doc A 16x32 layout
        af[ms].u[r] = *(const unsigned int*)(src + kk);
      }
    }
#pragma unroll
    for (int ns = 0; ns < 2; ++ns) {
      const unsigned short* src = &ldsB[cur][waveN + ns*16 + lm];
#pragma unroll
      for (int r = 0; r < 8; ++r) {
        const int kk = ((r & 4) << 2) + lh * 8 + (r & 3) * 2;
        bfr[ns].u[r] = (unsigned int)src[kk * LDB] | ((unsigned int)src[(kk + 1) * LDB] << 16);
      }
    }
#pragma unroll
    for (int ms = 0; ms < 4; ++ms)
#pragma unroll
      for (int ns = 0; ns < 2; ++ns)
        acc[ms][ns] = __builtin_amdgcn_wmma_f32_16x16x32_bf16(
            false, af[ms].v, false, bfr[ns].v, (short)0, acc[ms][ns], false, false);

    // ---- convert + stage tile k+1 into the other buffer
    if (more) {
      unsigned int* d = (unsigned int*)&ldsA[cur ^ 1][arow * LDA + akoff];
      d[0] = pk2(a0.x, a0.y); d[1] = pk2(a0.z, a0.w);
      d[2] = pk2(a1.x, a1.y); d[3] = pk2(a1.z, a1.w);
      d[4] = pk2(a2.x, a2.y); d[5] = pk2(a2.z, a2.w);
      d[6] = pk2(a3.x, a3.y); d[7] = pk2(a3.z, a3.w);
      unsigned short* db = &ldsB[cur ^ 1][akoff * LDB + arow];
      db[ 0*LDB] = f2bf(w0.x); db[ 1*LDB] = f2bf(w0.y);
      db[ 2*LDB] = f2bf(w0.z); db[ 3*LDB] = f2bf(w0.w);
      db[ 4*LDB] = f2bf(w1.x); db[ 5*LDB] = f2bf(w1.y);
      db[ 6*LDB] = f2bf(w1.z); db[ 7*LDB] = f2bf(w1.w);
      db[ 8*LDB] = f2bf(w2.x); db[ 9*LDB] = f2bf(w2.y);
      db[10*LDB] = f2bf(w2.z); db[11*LDB] = f2bf(w2.w);
      db[12*LDB] = f2bf(w3.x); db[13*LDB] = f2bf(w3.y);
      db[14*LDB] = f2bf(w3.z); db[15*LDB] = f2bf(w3.w);
    }
    __syncthreads();   // protects buf cur^1 writes vs next-iter reads
  }

#pragma unroll
  for (int ms = 0; ms < 4; ++ms)
#pragma unroll
    for (int ns = 0; ns < 2; ++ns)
#pragma unroll
      for (int j = 0; j < 8; ++j) {
        const int row = blockM + waveM + ms*16 + j + lh*8;     // doc C layout
        const int col = blockN + waveN + ns*16 + lm;
        C[(size_t)row * N + col] = acc[ms][ns][j];
      }
}

// ---------------------------------------------------------------------------
// Phase B: temporal IIR filter; drive[t] = ma_t + mb_t written over apical.
// ---------------------------------------------------------------------------
__global__ __launch_bounds__(256) void drive_kernel(
    float* __restrict__ apical,            // [T][2048][512], overwritten
    const float* __restrict__ basalC)      // [T][64][512]
{
  const size_t idx = (size_t)blockIdx.x * 256 + threadIdx.x;   // < 2048*512
  const int f = (int)(idx & 511);
  const int r = (int)(idx >> 9);
  const int b = r >> 5;                                        // row -> batch
  float ma = 0.f, mb = 0.f;
#pragma unroll
  for (int t = 0; t < TT; ++t) {
    const float ap = apical[(size_t)t * (NROW * FF) + idx];
    const float ba = basalC[((size_t)t * 64 + b) * FF + f];
    ma += (ap - ma) * TAU_INV;
    mb += (ba - mb) * TAU_INV;
    apical[(size_t)t * (NROW * FF) + idx] = ma + mb;
  }
}

// W1 -> bf16 transposed [k][n]; W2 -> bf16 transposed+padded [k][32]
__global__ __launch_bounds__(256) void convert_weights(
    const float* __restrict__ W1, const float* __restrict__ W2,
    unsigned short* __restrict__ w1t, unsigned short* __restrict__ w2t)
{
  const int idx = blockIdx.x * 256 + threadIdx.x;              // < 512*512
  const int k = idx >> 9, n = idx & 511;
  w1t[k * FF + n] = f2bf(W1[n * FF + k]);
  if (idx < LPAD * FF) {
    const int k2 = idx >> 5, n2 = idx & 31;
    w2t[k2 * LPAD + n2] = (n2 < LL) ? f2bf(W2[n2 * FF + k2]) : (unsigned short)0;
  }
}

// ---------------------------------------------------------------------------
// Phase C: sequential SNN. One block owns 16 rows for all timesteps.
// ---------------------------------------------------------------------------
#define ROWS 16
#define FS   520   // padded LDS stride

__global__ __launch_bounds__(256) void snn_kernel(
    const float* __restrict__ drive,           // [T][2048][512]
    const unsigned short* __restrict__ w1t,    // [512][512] k-major bf16
    const unsigned short* __restrict__ w2t,    // [512][32]  k-major bf16
    const float* __restrict__ b1,
    const float* __restrict__ b2,
    float* __restrict__ out)                   // [64][18][32]
{
  extern __shared__ char smem[];
  float*          msL  = (float*)smem;                          // 16*520 f32
  unsigned short* spA  = (unsigned short*)(smem + ROWS*FS*4);   // 16*520 bf16
  unsigned short* sp2A = spA + ROWS*FS;                         // 16*520 bf16
  float*          outL = (float*)(sp2A + ROWS*FS);              // 16*32 f32

  const int tid  = threadIdx.x;
  const int wave = tid >> 5;
  const int lane = tid & 31;
  const int lm = lane & 15, lh = lane >> 4;
  const int row0 = blockIdx.x * ROWS;

  for (int i = tid; i < ROWS * FS; i += 256) msL[i] = 0.f;

  float b1r[4];
#pragma unroll
  for (int ns = 0; ns < 4; ++ns) b1r[ns] = b1[wave*64 + ns*16 + lm];

  const v8f vzero = {0.f,0.f,0.f,0.f,0.f,0.f,0.f,0.f};
  v8f mlr[4], outacc[2];
#pragma unroll
  for (int i = 0; i < 4; ++i) mlr[i] = vzero;
#pragma unroll
  for (int i = 0; i < 2; ++i) outacc[i] = vzero;

  const int urow = tid >> 4;           // 0..15
  const int uf0  = (tid & 15) * 32;    // 32 consecutive f per thread
  __syncthreads();

  for (int t = 0; t < TT; ++t) {
    // -- phase 1: soma membrane integrate + spike + hard reset
    const float* dp = drive + ((size_t)t * NROW + row0 + urow) * FF + uf0;
#pragma unroll
    for (int i = 0; i < 32; i += 4) {
      const float4 d = *(const float4*)(dp + i);
      const float dv[4] = {d.x, d.y, d.z, d.w};
#pragma unroll
      for (int c = 0; c < 4; ++c) {
        const int fidx = urow * FS + uf0 + i + c;
        float m = msL[fidx];
        m += (dv[c] - m) * TAU_INV;
        const float s = (m > 1.0f) ? 1.0f : 0.0f;
        msL[fidx] = m * (1.0f - s);
        spA[fidx] = (s > 0.f) ? (unsigned short)0x3F80u : (unsigned short)0u;
      }
    }
    __syncthreads();

    // -- phase 2: h = sp @ W1^T (M=16,N=512,K=512); each wave 4 N-tiles
    v8f acc1[4];
#pragma unroll
    for (int i = 0; i < 4; ++i) acc1[i] = vzero;
    for (int kt = 0; kt < FF; kt += 32) {
      Frag a;
      const unsigned short* as = &spA[lm * FS + kt];
#pragma unroll
      for (int r = 0; r < 8; ++r) {
        const int kk = ((r & 4) << 2) + lh * 8 + (r & 3) * 2;
        a.u[r] = *(const unsigned int*)(as + kk);
      }
#pragma unroll
      for (int ns = 0; ns < 4; ++ns) {
        Frag bb;
        const unsigned short* bs = &w1t[(size_t)kt * FF + wave*64 + ns*16 + lm];
#pragma unroll
        for (int r = 0; r < 8; ++r) {
          const int kk = ((r & 4) << 2) + lh * 8 + (r & 3) * 2;
          bb.u[r] = (unsigned int)bs[(size_t)kk * FF] | ((unsigned int)bs[(size_t)(kk + 1) * FF] << 16);
        }
        acc1[ns] = __builtin_amdgcn_wmma_f32_16x16x32_bf16(
            false, a.v, false, bb.v, (short)0, acc1[ns], false, false);
      }
    }
    // LIF on ml (registers, fragment-aligned), emit sp2 to LDS
#pragma unroll
    for (int ns = 0; ns < 4; ++ns)
#pragma unroll
      for (int j = 0; j < 8; ++j) {
        const float h = acc1[ns][j] + b1r[ns];
        float m = mlr[ns][j];
        m += (h - m) * TAU_INV;
        const float s2 = (m > 0.5f) ? 1.0f : 0.0f;
        mlr[ns][j] = m * (1.0f - s2);
        const int rj  = j + lh * 8;
        const int col = wave*64 + ns*16 + lm;
        sp2A[rj * FS + col] = (s2 > 0.f) ? (unsigned short)0x3F80u
                                         : (unsigned short)0u;
      }
    __syncthreads();

    // -- phase 3: out += sp2 @ W2^T, K split across waves (64 k each)
#pragma unroll
    for (int kt2 = 0; kt2 < 2; ++kt2) {
      const int k0 = wave * 64 + kt2 * 32;
      Frag a2;
      const unsigned short* as = &sp2A[lm * FS + k0];
#pragma unroll
      for (int r = 0; r < 8; ++r) {
        const int kk = ((r & 4) << 2) + lh * 8 + (r & 3) * 2;
        a2.u[r] = *(const unsigned int*)(as + kk);
      }
#pragma unroll
      for (int ns = 0; ns < 2; ++ns) {
        Frag bb;
        const unsigned short* bs = &w2t[(size_t)k0 * LPAD + ns*16 + lm];
#pragma unroll
        for (int r = 0; r < 8; ++r) {
          const int kk = ((r & 4) << 2) + lh * 8 + (r & 3) * 2;
          bb.u[r] = (unsigned int)bs[kk * LPAD] | ((unsigned int)bs[(kk + 1) * LPAD] << 16);
        }
        outacc[ns] = __builtin_amdgcn_wmma_f32_16x16x32_bf16(
            false, a2.v, false, bb.v, (short)0, outacc[ns], false, false);
      }
    }
    __syncthreads();
  }

  // -- cross-wave K reduction via LDS float atomics, then store [B,L,S]
  outL[tid]       = 0.f;
  outL[tid + 256] = 0.f;
  __syncthreads();
#pragma unroll
  for (int ns = 0; ns < 2; ++ns)
#pragma unroll
    for (int j = 0; j < 8; ++j)
      atomicAdd(&outL[(j + lh*8) * LPAD + ns*16 + lm], outacc[ns][j]);
  __syncthreads();
  if (tid < ROWS * LL) {
    const int r = tid / LL, l = tid % LL;
    const int gr = row0 + r;
    const int b = gr >> 5, s = gr & 31;
    out[((size_t)b * LL + l) * SS + s] = outL[r * LPAD + l] * (1.0f / (float)TT)
                                         + b2[l];
  }
}

// ---------------------------------------------------------------------------
extern "C" void kernel_launch(void* const* d_in, const int* in_sizes, int n_in,
                              void* d_out, int out_size, void* d_ws, size_t ws_size,
                              hipStream_t stream) {
  (void)in_sizes; (void)n_in; (void)out_size; (void)ws_size;
  const float* se = (const float*)d_in[0];   // [8,64,3136]
  const float* te = (const float*)d_in[1];   // [8,2048,3136]
  const float* Wb = (const float*)d_in[2];   // [512,3136]
  const float* Wa = (const float*)d_in[3];   // [512,3136]
  const float* W1 = (const float*)d_in[4];   // [512,512]
  const float* b1 = (const float*)d_in[5];   // [512]
  const float* W2 = (const float*)d_in[6];   // [18,512]
  const float* b2 = (const float*)d_in[7];   // [18]
  float* out = (float*)d_out;                // [64,18,32]

  char* ws = (char*)d_ws;
  float* apical = (float*)ws;                                  // 8*2048*512 f32 (becomes drive)
  float* basalC = apical + (size_t)TT * NROW * FF;             // 8*64*512 f32
  unsigned short* w1t = (unsigned short*)(basalC + (size_t)TT * 64 * FF);
  unsigned short* w2t = w1t + FF * FF;

  // time-batched projections
  gemm_f32_bf16<<<dim3(FF / BN, (TT * NROW) / BM), 256, 0, stream>>>(
      te, Wa, apical, TT * NROW, FF, 3136);
  gemm_f32_bf16<<<dim3(FF / BN, (TT * 64) / BM), 256, 0, stream>>>(
      se, Wb, basalC, TT * 64, FF, 3136);
  // ma/mb linear filters -> drive[t] (in place)
  drive_kernel<<<(NROW * FF) / 256, 256, 0, stream>>>(apical, basalC);
  // weight precompute
  convert_weights<<<(FF * FF) / 256, 256, 0, stream>>>(W1, W2, w1t, w2t);
  // sequential SNN (dynamic LDS: 16*520*(4+2+2) + 16*32*4 = 68608 B)
  snn_kernel<<<NROW / ROWS, 256, 68608, stream>>>(apical, w1t, w2t, b1, b2, out);
}
